// ASAP_18657337933841
// MI455X (gfx1250) — matrix operand order repaired and checked by reading, not compiled, not run on hardware
//
#include <hip/hip_runtime.h>
#include <hip/hip_bf16.h>
#include <math.h>

typedef __attribute__((ext_vector_type(16))) _Float16 v16h;
typedef __attribute__((ext_vector_type(8)))  _Float16 v8h;
typedef __attribute__((ext_vector_type(8)))  float    v8f;
typedef __attribute__((ext_vector_type(4)))  unsigned int u32x4;
typedef __attribute__((ext_vector_type(8)))  int          i32x8;
typedef __attribute__((ext_vector_type(4)))  int          i32x4;

#define NTOT   51200      // B*N
#define NGRAPH 256
#define NPG    200        // nodes per graph
#define HDIM   128
#define PANK   32         // K-depth of one staged W panel
#define PSTR   40         // LDS row stride in f16 (32 + 8 pad) -> conflict-free, 16B aligned
#define PELEM  (HDIM * PSTR)   // f16 elements per panel (5120)

// ---------------- utility kernels ----------------

__global__ void bg_zero_f32(float* __restrict__ p, long n) {
    long i = (long)blockIdx.x * blockDim.x + threadIdx.x;
    if (i < n) p[i] = 0.0f;
}

__global__ void bg_degree(const int* __restrict__ dst, float* __restrict__ deg, int E) {
    int e = blockIdx.x * blockDim.x + threadIdx.x;
    if (e < E) atomicAdd(&deg[dst[e]], 1.0f);
}

// one block per edge; thread f adds feature f
__global__ void bg_scatter_f32(const float* __restrict__ x, const int* __restrict__ src,
                               const int* __restrict__ dst, float* __restrict__ agg, int F) {
    int e = blockIdx.x;
    int f = threadIdx.x;
    if (f < F) {
        int s = src[e], d = dst[e];
        atomicAdd(&agg[(size_t)d * F + f], x[(size_t)s * F + f]);
    }
}

__global__ void bg_scatter_f16(const _Float16* __restrict__ x, const int* __restrict__ src,
                               const int* __restrict__ dst, float* __restrict__ agg, int F) {
    int e = blockIdx.x;
    int f = threadIdx.x;
    if (f < F) {
        int s = src[e], d = dst[e];
        atomicAdd(&agg[(size_t)d * F + f], (float)x[(size_t)s * F + f]);
    }
}

// mean = agg/max(deg,1), cast to f16, zero-pad K..KP  (grid=NT, block=KP)
__global__ void bg_mean_f16(const float* __restrict__ agg, const float* __restrict__ deg,
                            _Float16* __restrict__ outp, int K, int KP) {
    int i = blockIdx.x, f = threadIdx.x;
    float d = deg[i];
    float r = 1.0f / (d > 1.0f ? d : 1.0f);
    float v = (f < K) ? agg[(size_t)i * K + f] * r : 0.0f;
    outp[(size_t)i * KP + f] = (_Float16)v;
}

// f32 -> f16 with zero pad  (grid=NT, block=KP)
__global__ void bg_cvt_f16(const float* __restrict__ x, _Float16* __restrict__ outp,
                           int K, int KP) {
    int i = blockIdx.x, f = threadIdx.x;
    float v = (f < K) ? x[(size_t)i * K + f] : 0.0f;
    outp[(size_t)i * KP + f] = (_Float16)v;
}

// W [K,128] f32  ->  wt [128, KP] f16 (transposed, zero-padded). grid=128, block=KP
__global__ void bg_prep_w(const float* __restrict__ W, _Float16* __restrict__ wt,
                          int K, int KP) {
    int n = blockIdx.x, k = threadIdx.x;
    float v = (k < K) ? W[(size_t)k * HDIM + n] : 0.0f;
    wt[(size_t)n * KP + k] = (_Float16)v;
}

// ---------------- TDM helper: DMA one 32x128 W panel into LDS ----------------
// Loads tile [tile_dim0=32 k, tile_dim1=128 n] of an f16 tensor with row stride KP,
// writing rows into LDS with hardware padding: pad 4 DWORDs after every 64 bytes
// stored  =>  LDS row stride = 40 f16 (= PSTR).
__device__ __forceinline__ void tdm_load_w_panel(unsigned int lds_off,
                                                 const _Float16* wt, int kp, int KP) {
    unsigned long long ga = (unsigned long long)(size_t)wt + (unsigned long long)kp * 2ull;
    u32x4 g0;
    g0[0] = 1u;                                        // count=1 valid descriptor
    g0[1] = lds_off;                                   // LDS byte address
    g0[2] = (unsigned int)(ga & 0xffffffffull);        // global_addr[31:0]
    g0[3] = (unsigned int)((ga >> 32) & 0x1ffffffull)  // global_addr[56:32]
          | (2u << 30);                                // type = 2 ("image")
    i32x8 g1;
    g1[0] = (int)((1u << 16)      // data_size = 2 bytes
                | (1u << 20)      // pad_enable
                | (3u << 22)      // pad_interval: after 2^3 * 8B = 64B (= 32 f16 row)
                | (3u << 25));    // pad_amount: 4 DWORDs (= 8 f16)  -> stride 40 f16
    g1[1] = (int)(((unsigned)KP & 0xffffu) << 16);     // tensor_dim0[15:0]
    g1[2] = (int)(128u << 16);                         // tensor_dim1[15:0] = 128
    g1[3] = (int)((unsigned)PANK << 16);               // tile_dim0 = 32
    g1[4] = (int)128;                                  // tile_dim1 = 128
    g1[5] = (int)KP;                                   // tensor_dim0_stride[31:0]
    g1[6] = 0;
    g1[7] = 0;
    i32x4 z4 = {0, 0, 0, 0};
#if __clang_major__ >= 23
    i32x8 z8 = {0, 0, 0, 0, 0, 0, 0, 0};
    __builtin_amdgcn_tensor_load_to_lds(g0, g1, z4, z4, z8, 0);
#else
    __builtin_amdgcn_tensor_load_to_lds(g0, g1, z4, z4, 0);
#endif
}

// ---------------- WMMA dual GEMM:  out = relu(A1@W1t + A2@W2t + bias) ----------------
// A1,A2: [NT, KP] f16 (zero-padded). wt1,wt2: [128, KP] f16 transposed weights.
// block = 256 thr = 8 waves; wave w owns rows [blk*128 + w*16, +16), all 128 cols.
// W panels are double-buffered in LDS, filled by the Tensor Data Mover; B fragments
// are software-pipelined (prefetch tile nt+1 while WMMA consumes tile nt).
__global__ __launch_bounds__(256)
void bg_gemm_dual_relu(const _Float16* __restrict__ A1, const _Float16* __restrict__ A2,
                       const _Float16* __restrict__ wt1, const _Float16* __restrict__ wt2,
                       const float* __restrict__ bias, _Float16* __restrict__ outp,
                       int K, int KP) {
    __shared__ _Float16 ldsW[2][2][PELEM];   // [buffer][matrix][n*PSTR + k]

    const int tid  = threadIdx.x;
    const int wave = tid >> 5;
    const int lane = tid & 31;
    const int nloc = lane & 15;
    const int half = lane >> 4;
    const int m0   = blockIdx.x * 128 + wave * 16;
    const int rowA = m0 + nloc;       // A-matrix row this lane loads
    const int c0   = half * 8;        // A k-chunk base within a 32-wide K step
    const int K0   = half * 8;        // B k-chunk base within a 32-wide K step

    v8f acc[8];
    #pragma unroll
    for (int i = 0; i < 8; ++i)
        #pragma unroll
        for (int j = 0; j < 8; ++j) acc[i][j] = 0.0f;

    const int NP = (K + PANK - 1) / PANK;   // panels of 32 K (A/W zero-padded to KP)

    // LDS byte offsets of the four panel buffers (generic ptr low 32 bits = LDS offset)
    unsigned int ldsOff[2][2];
    #pragma unroll
    for (int b = 0; b < 2; ++b)
        #pragma unroll
        for (int m = 0; m < 2; ++m)
            ldsOff[b][m] = (unsigned int)(size_t)(&ldsW[b][m][0]);

    // prologue: DMA panel 0 into buffer 0
    if (wave == 0) {
        tdm_load_w_panel(ldsOff[0][0], wt1, 0, KP);
        tdm_load_w_panel(ldsOff[0][1], wt2, 0, KP);
    }

    for (int p = 0; p < NP; ++p) {
        const int buf = p & 1;
        if (wave == 0) {
            if (p + 1 < NP) {
                // prefetch next panel into the other buffer, then wait for panel p
                tdm_load_w_panel(ldsOff[buf ^ 1][0], wt1, (p + 1) * PANK, KP);
                tdm_load_w_panel(ldsOff[buf ^ 1][1], wt2, (p + 1) * PANK, KP);
                __builtin_amdgcn_s_wait_tensorcnt(2);
            } else {
                __builtin_amdgcn_s_wait_tensorcnt(0);
            }
        }
        __syncthreads();   // panel p visible to all waves

        const int kp = p * PANK;
        // A fragments: 16-bit A 16x32 layout -> two 16B chunks per lane
        const _Float16* ap1 = A1 + (size_t)rowA * KP + kp + c0;
        const _Float16* ap2 = A2 + (size_t)rowA * KP + kp + c0;
        v8h lo1 = *(const v8h*)ap1;
        v8h hi1 = *(const v8h*)(ap1 + 16);
        v8h lo2 = *(const v8h*)ap2;
        v8h hi2 = *(const v8h*)(ap2 + 16);
        v16h a1, a2;
        #pragma unroll
        for (int j = 0; j < 8; ++j) {
            a1[j] = lo1[j]; a1[j + 8] = hi1[j];
            a2[j] = lo2[j]; a2[j + 8] = hi2[j];
        }

        // two passes (matrix 0 with a1, matrix 1 with a2); B fragments prefetched
        // one N-tile ahead so each WMMA has a full tile of slack over its ds_loads.
        #pragma unroll
        for (int pass = 0; pass < 2; ++pass) {
            const v16h a = pass ? a2 : a1;
            const _Float16* bb = &ldsW[buf][pass][nloc * PSTR + K0];
            v8h blo = *(const v8h*)bb;
            v8h bhi = *(const v8h*)(bb + 16);
            #pragma unroll
            for (int nt = 0; nt < 8; ++nt) {
                v8h nblo, nbhi;
                if (nt < 7) {
                    const _Float16* bn = bb + (nt + 1) * (16 * PSTR);
                    nblo = *(const v8h*)bn;
                    nbhi = *(const v8h*)(bn + 16);
                }
                v16h bfr;
                #pragma unroll
                for (int j = 0; j < 8; ++j) {
                    bfr[j] = blo[j]; bfr[j + 8] = bhi[j];
                }
                acc[nt] = __builtin_amdgcn_wmma_f32_16x16x32_f16(
                    false, a, false, bfr, (short)0, acc[nt], false, false);
                blo = nblo; bhi = nbhi;
            }
        }
        __syncthreads();   // all waves done reading buf before it is DMA-overwritten
    }

    // epilogue: C/D layout -> lane holds col n = nt*16+nloc, rows m0 + vr + 8*half
    #pragma unroll
    for (int nt = 0; nt < 8; ++nt) {
        const int n = nt * 16 + nloc;
        const float bv = bias[n];
        #pragma unroll
        for (int vr = 0; vr < 8; ++vr) {
            int row = m0 + vr + 8 * half;
            float v = acc[nt][vr] + bv;
            v = v > 0.0f ? v : 0.0f;
            outp[(size_t)row * HDIM + n] = (_Float16)v;
        }
    }
}

// ---------------- pooling / MLP head ----------------

// grid = NGRAPH, block = 128; contiguous 200-row blocks per graph
__global__ void bg_pool(const _Float16* __restrict__ h, float* __restrict__ z,
                        int zcol, float scale) {
    int b = blockIdx.x, n = threadIdx.x;
    const _Float16* p = h + (size_t)b * NPG * HDIM + n;
    float s = 0.0f;
    for (int i = 0; i < NPG; ++i) s += (float)p[(size_t)i * HDIM];
    z[b * 512 + zcol + n] = s * scale;
}

__global__ void bg_linear_relu(const float* __restrict__ in, const float* __restrict__ W,
                               const float* __restrict__ b, float* __restrict__ outp,
                               int M, int K, int N) {
    int t = blockIdx.x * blockDim.x + threadIdx.x;
    if (t >= M * N) return;
    int m = t / N, n = t % N;
    float s = b[n];
    for (int k = 0; k < K; ++k) s += in[(size_t)m * K + k] * W[(size_t)k * N + n];
    outp[(size_t)m * N + n] = s > 0.0f ? s : 0.0f;
}

// lin3 (64->2) + log_softmax over 2 classes; one thread per row
__global__ void bg_head(const float* __restrict__ in, const float* __restrict__ W,
                        const float* __restrict__ b, float* __restrict__ outp) {
    int m = blockIdx.x * blockDim.x + threadIdx.x;
    if (m >= NGRAPH) return;
    float o0 = b[0], o1 = b[1];
    for (int k = 0; k < 64; ++k) {
        float v = in[(size_t)m * 64 + k];
        o0 += v * W[k * 2 + 0];
        o1 += v * W[k * 2 + 1];
    }
    float mx  = o0 > o1 ? o0 : o1;
    float lse = mx + logf(expf(o0 - mx) + expf(o1 - mx));
    outp[m * 2 + 0] = o0 - lse;
    outp[m * 2 + 1] = o1 - lse;
}

// ---------------- driver ----------------

extern "C" void kernel_launch(void* const* d_in, const int* in_sizes, int n_in,
                              void* d_out, int out_size, void* d_ws, size_t ws_size,
                              hipStream_t stream) {
    const int E = in_sizes[2] / 2;   // 819200

    const float* xin[2]  = { (const float*)d_in[0], (const float*)d_in[1] };
    const int*   ei[2]   = { (const int*)d_in[2],   (const int*)d_in[3]   };

    // workspace carve-out (256B aligned)
    char* Wb = (char*)d_ws;
    size_t off = 0;
    auto nxt = [&](size_t bytes) -> void* {
        void* p = Wb + off;
        off = (off + bytes + 255) & ~(size_t)255;
        return p;
    };
    float*    agg   = (float*)   nxt((size_t)NTOT * 200 * 4);
    float*    deg   = (float*)   nxt((size_t)NTOT * 4);
    _Float16* amean = (_Float16*)nxt((size_t)NTOT * 256 * 2);
    _Float16* aroot = (_Float16*)nxt((size_t)NTOT * 256 * 2);
    _Float16* h1    = (_Float16*)nxt((size_t)NTOT * 128 * 2);
    _Float16* h2    = (_Float16*)nxt((size_t)NTOT * 128 * 2);
    _Float16* wt1   = (_Float16*)nxt((size_t)HDIM * 256 * 2);
    _Float16* wt2   = (_Float16*)nxt((size_t)HDIM * 256 * 2);
    float*    z     = (float*)   nxt((size_t)NGRAPH * 512 * 4);
    float*    z2    = (float*)   nxt((size_t)NGRAPH * 128 * 4);
    float*    z3    = (float*)   nxt((size_t)NGRAPH * 64 * 4);

    for (int br = 0; br < 2; ++br) {
        const float* x   = xin[br];
        const int*   src = ei[br];
        const int*   dst = src + E;
        const int    wb  = 5 + br * 6;
        const float* w1rel  = (const float*)d_in[wb + 0];
        const float* b1rel  = (const float*)d_in[wb + 1];
        const float* w1root = (const float*)d_in[wb + 2];
        const float* w2rel  = (const float*)d_in[wb + 3];
        const float* b2rel  = (const float*)d_in[wb + 4];
        const float* w2root = (const float*)d_in[wb + 5];
        const int   zcol  = br * 256;
        const float scale = (br == 0) ? (1.0f / (float)NPG) : 1.0f;   // sc: mean, fc: sum

        // ---- conv1 (K=200, KP=256) ----
        bg_zero_f32<<<((long)NTOT * 200 + 255) / 256, 256, 0, stream>>>(agg, (long)NTOT * 200);
        bg_zero_f32<<<(NTOT + 255) / 256, 256, 0, stream>>>(deg, NTOT);
        bg_degree<<<(E + 255) / 256, 256, 0, stream>>>(dst, deg, E);
        bg_scatter_f32<<<E, 256, 0, stream>>>(x, src, dst, agg, 200);
        bg_mean_f16<<<NTOT, 256, 0, stream>>>(agg, deg, amean, 200, 256);
        bg_cvt_f16<<<NTOT, 256, 0, stream>>>(x, aroot, 200, 256);
        bg_prep_w<<<HDIM, 256, 0, stream>>>(w1rel,  wt1, 200, 256);
        bg_prep_w<<<HDIM, 256, 0, stream>>>(w1root, wt2, 200, 256);
        bg_gemm_dual_relu<<<NTOT / 128, 256, 0, stream>>>(amean, aroot, wt1, wt2,
                                                          b1rel, h1, 200, 256);
        bg_pool<<<NGRAPH, 128, 0, stream>>>(h1, z, zcol + 0, scale);

        // ---- conv2 (K=128, KP=128), same edges -> same degrees ----
        bg_zero_f32<<<((long)NTOT * 128 + 255) / 256, 256, 0, stream>>>(agg, (long)NTOT * 128);
        bg_scatter_f16<<<E, 128, 0, stream>>>(h1, src, dst, agg, 128);
        bg_mean_f16<<<NTOT, 128, 0, stream>>>(agg, deg, amean, 128, 128);
        bg_prep_w<<<HDIM, 128, 0, stream>>>(w2rel,  wt1, 128, 128);
        bg_prep_w<<<HDIM, 128, 0, stream>>>(w2root, wt2, 128, 128);
        bg_gemm_dual_relu<<<NTOT / 128, 256, 0, stream>>>(amean, h1, wt1, wt2,
                                                          b2rel, h2, 128, 128);
        bg_pool<<<NGRAPH, 128, 0, stream>>>(h2, z, zcol + 128, scale);
    }

    // ---- MLP head ----
    const float* lin1_w = (const float*)d_in[17];
    const float* lin1_b = (const float*)d_in[18];
    const float* lin2_w = (const float*)d_in[19];
    const float* lin2_b = (const float*)d_in[20];
    const float* lin3_w = (const float*)d_in[21];
    const float* lin3_b = (const float*)d_in[22];

    bg_linear_relu<<<(NGRAPH * 128 + 255) / 256, 256, 0, stream>>>(z,  lin1_w, lin1_b, z2,
                                                                   NGRAPH, 512, 128);
    bg_linear_relu<<<(NGRAPH * 64 + 255) / 256, 256, 0, stream>>>(z2, lin2_w, lin2_b, z3,
                                                                  NGRAPH, 128, 64);
    bg_head<<<1, 256, 0, stream>>>(z3, lin3_w, lin3_b, (float*)d_out);
}
